// RawAttention_44667659878694
// MI455X (gfx1250) — compile-verified, matrix operand order checked by reading, not run on hardware
//
#include <hip/hip_runtime.h>
#include <hip/hip_bf16.h>

typedef __attribute__((ext_vector_type(16))) _Float16 v16h;
typedef __attribute__((ext_vector_type(8)))  _Float16 h8;
typedef __attribute__((ext_vector_type(4)))  _Float16 h4;
typedef __attribute__((ext_vector_type(8)))  float    v8f;
typedef __attribute__((ext_vector_type(4)))  float    f4;

#define B_ 4
#define L_ 2048
#define H_ 16
#define E_ 64

// LDS row pitches (halves), padded for bank spread, 16B-aligned
#define KP 72   // K tile: 32 n-rows x 64 e, pitch 144B
#define VP 40   // V^T tile: 64 d-rows x 32 n, pitch 80B

// ---------------- Pre-pass 1: K (B,S,H,E) f32 -> Kh (B,H,S,E) f16 ----------------
__global__ __launch_bounds__(256)
void conv_k_kernel(const float* __restrict__ K, _Float16* __restrict__ Kh) {
    const size_t q = (size_t)blockIdx.x * 256 + threadIdx.x;  // quad id, 2^21 total
    const int e4 = (int)(q & 15);
    const int h  = (int)((q >> 4) & 15);
    const int s  = (int)((q >> 8) & 2047);
    const int b  = (int)(q >> 19);
    f4 v = *(const f4*)(K + q * 4);
    h4 o;
    o[0] = (_Float16)v[0]; o[1] = (_Float16)v[1];
    o[2] = (_Float16)v[2]; o[3] = (_Float16)v[3];
    *(h4*)(Kh + (((size_t)(b * H_ + h) * L_ + s) * E_) + 4 * e4) = o;
}

// ---------------- Pre-pass 2: V (B,S,H,D) f32 -> Vt (B,H,D,S) f16 ----------------
#define TP 68   // transpose tile pitch (halves), 8B-aligned rows
__global__ __launch_bounds__(256)
void trans_v_kernel(const float* __restrict__ V, _Float16* __restrict__ Vt) {
    __shared__ __attribute__((aligned(16))) _Float16 T[64 * TP];
    const int tid = threadIdx.x;
    const int wg  = blockIdx.x;               // B*H*(L/64) = 2048
    const int st  = wg & 31;                  // s-tile
    const int bh  = wg >> 5;
    const int h   = bh & 15;
    const int b   = bh >> 4;
    const int s0  = st * 64;

    #pragma unroll
    for (int it = 0; it < 4; ++it) {          // load 64x64 f32 tile, convert
        const int p  = tid + 256 * it;
        const int si = p >> 4;
        const int d4 = (p & 15) << 2;
        f4 v = *(const f4*)(V + (((size_t)b * L_ + s0 + si) * H_ + h) * E_ + d4);
        h4 o;
        o[0] = (_Float16)v[0]; o[1] = (_Float16)v[1];
        o[2] = (_Float16)v[2]; o[3] = (_Float16)v[3];
        *(h4*)&T[si * TP + d4] = o;
    }
    __syncthreads();
    #pragma unroll
    for (int it = 0; it < 2; ++it) {          // write transposed rows (d, s-contig)
        const int p   = tid + 256 * it;
        const int d   = p >> 3;
        const int sub = p & 7;
        h8 o;
        #pragma unroll
        for (int k = 0; k < 8; ++k) o[k] = T[(8 * sub + k) * TP + d];
        *(h8*)(Vt + ((size_t)(b * H_ + h) * E_ + d) * L_ + s0 + 8 * sub) = o;
    }
}

// ---------------- Main flash-attention kernel ----------------
__global__ __launch_bounds__(256)
void fa_causal_wmma_kernel(const float* __restrict__ Qg,
                           const _Float16* __restrict__ Kh,
                           const _Float16* __restrict__ Vt,
                           float* __restrict__ Og) {
    __shared__ __attribute__((aligned(16))) _Float16 Klds[2][32 * KP];
    __shared__ __attribute__((aligned(16))) _Float16 Vlds[2][64 * VP];

    const int tid  = threadIdx.x;
    const int lane = tid & 31;
    const int wave = tid >> 5;
    const int hi   = (lane >> 4) & 1;
    const int lm   = lane & 15;

    const int wg   = blockIdx.x;
    const int qblk = wg & 15;
    const int bh   = wg >> 4;
    const int h    = bh & 15;
    const int b    = bh >> 4;

    const int qb = qblk * 128;
    const int q0 = qb + wave * 16;
    const int nchunks = (qb >> 5) + 4;

    const _Float16* Kbh = Kh + (size_t)(b * H_ + h) * L_ * E_;  // [s][e]
    const _Float16* Vbh = Vt + (size_t)(b * H_ + h) * E_ * L_;  // [d][s]

    // this thread's fixed staging assignment (one b128 per tensor per chunk)
    const int kn = tid >> 3, ks = tid & 7;   // K: row n, 16B piece
    const int vd = tid >> 2, vs = tid & 3;   // V: row d, 16B piece

    // ---- Q tile -> two f16 B-operands, scale * log2(e) folded in ----
    const float qfac = 0.125f * 1.44269504088896340736f;
    const float* qrow = Qg + (((size_t)b * L_ + (size_t)(q0 + lm)) * H_ + h) * E_;
    v16h qB[2];
    #pragma unroll
    for (int eh = 0; eh < 2; ++eh) {
        const int e0 = 32 * eh + 8 * hi;
        f4 a0 = *(const f4*)(qrow + e0);
        f4 a1 = *(const f4*)(qrow + e0 + 4);
        f4 b0 = *(const f4*)(qrow + e0 + 16);
        f4 b1 = *(const f4*)(qrow + e0 + 20);
        v16h t;
        #pragma unroll
        for (int i = 0; i < 4; ++i) {
            t[i]      = (_Float16)(a0[i] * qfac);
            t[4 + i]  = (_Float16)(a1[i] * qfac);
            t[8 + i]  = (_Float16)(b0[i] * qfac);
            t[12 + i] = (_Float16)(b1[i] * qfac);
        }
        qB[eh] = t;
    }

    v8f acc[4];
    #pragma unroll
    for (int j = 0; j < 4; ++j) acc[j] = (v8f){};
    float runM = -3.0e38f;
    float runS = 0.0f;

    // async stage of one 32-row K/V chunk into LDS buffer `buf`
    auto issue_chunk = [&](int c0, int buf) {
        {
            const unsigned long long ga =
                (unsigned long long)(const void*)(Kbh + (size_t)(c0 + kn) * E_ + ks * 8);
            const unsigned ld = (unsigned)(unsigned long long)(&Klds[buf][kn * KP + ks * 8]);
            asm volatile("global_load_async_to_lds_b128 %0, %1, off"
                         :: "v"(ld), "v"(ga) : "memory");
        }
        {
            const unsigned long long ga =
                (unsigned long long)(const void*)(Vbh + (size_t)vd * L_ + c0 + vs * 8);
            const unsigned ld = (unsigned)(unsigned long long)(&Vlds[buf][vd * VP + vs * 8]);
            asm volatile("global_load_async_to_lds_b128 %0, %1, off"
                         :: "v"(ld), "v"(ga) : "memory");
        }
    };

    issue_chunk(0, 0);                      // prologue: chunk 0 -> buf 0

    for (int c = 0; c < nchunks; ++c) {
        const int c0  = c * 32;
        const int cur = c & 1;

        if (c + 1 < nchunks) {
            issue_chunk((c + 1) * 32, cur ^ 1);
            asm volatile("s_wait_asynccnt 0x2" ::: "memory");  // chunk c landed
        } else {
            asm volatile("s_wait_asynccnt 0x0" ::: "memory");
        }
        __syncthreads();

        if (c0 <= q0 + 15) {
            const _Float16* Kb = &Klds[cur][0];
            const _Float16* Vb = &Vlds[cur][0];

            // ---- S^T = K_chunk * Q^T ----
            v8f ct[2];
            #pragma unroll
            for (int t = 0; t < 2; ++t) {
                v8f cacc = (v8f){};
                const int row = 16 * t + lm;
                #pragma unroll
                for (int eh = 0; eh < 2; ++eh) {
                    const int e0 = 32 * eh + 8 * hi;
                    h8 lo = *(const h8*)&Kb[row * KP + e0];
                    h8 hv = *(const h8*)&Kb[row * KP + e0 + 16];
                    v16h A = __builtin_shufflevector(lo, hv,
                        0, 1, 2, 3, 4, 5, 6, 7, 8, 9, 10, 11, 12, 13, 14, 15);
                    cacc = __builtin_amdgcn_wmma_f32_16x16x32_f16(
                        false, A, false, qB[eh], (short)0, cacc, false, false);
                }
                ct[t] = cacc;
            }

            // ---- causal mask near the diagonal ----
            if (c0 + 31 > q0) {
                const int lim0 = (q0 + lm) - (c0 + 8 * hi);
                #pragma unroll
                for (int v = 0; v < 8; ++v) {
                    if (v > lim0)      ct[0][v] = -3.0e38f;
                    if (v > lim0 - 16) ct[1][v] = -3.0e38f;
                }
            }

            // ---- online softmax (base-2) ----
            float cmax = ct[0][0];
            #pragma unroll
            for (int v = 1; v < 8; ++v) cmax = fmaxf(cmax, ct[0][v]);
            #pragma unroll
            for (int v = 0; v < 8; ++v) cmax = fmaxf(cmax, ct[1][v]);
            cmax = fmaxf(cmax, __shfl_xor(cmax, 16, 32));
            const float nm   = fmaxf(runM, cmax);
            const float corr = exp2f(runM - nm);
            runM = nm;

            v16h ap;
            float ps = 0.0f;
            #pragma unroll
            for (int v = 0; v < 8; ++v) {
                float p0 = exp2f(ct[0][v] - nm);
                float p1 = exp2f(ct[1][v] - nm);
                ps += p0 + p1;
                ap[v]     = (_Float16)p0;
                ap[8 + v] = (_Float16)p1;
            }
            ps += __shfl_xor(ps, 16, 32);
            runS = runS * corr + ps;

            v8f cs;
            #pragma unroll
            for (int v = 0; v < 8; ++v) cs[v] = __shfl(corr, v + 8 * hi, 32);
            #pragma unroll
            for (int j = 0; j < 4; ++j) acc[j] *= cs;

            // ---- O += P * V ----
            #pragma unroll
            for (int j = 0; j < 4; ++j) {
                const int d = 16 * j + lm;
                h8 lo = *(const h8*)&Vb[d * VP + 8 * hi];
                h8 hv = *(const h8*)&Vb[d * VP + 8 * hi + 16];
                v16h Bv = __builtin_shufflevector(lo, hv,
                    0, 1, 2, 3, 4, 5, 6, 7, 8, 9, 10, 11, 12, 13, 14, 15);
                acc[j] = __builtin_amdgcn_wmma_f32_16x16x32_f16(
                    false, ap, false, Bv, (short)0, acc[j], false, false);
            }
        }
        __syncthreads();
    }

    // ---- normalize + store ----
    #pragma unroll
    for (int v = 0; v < 8; ++v) {
        const float s    = __shfl(runS, v + 8 * hi, 32);
        const float rinv = 1.0f / s;
        const int   m    = v + 8 * hi;
        const size_t o   = (((size_t)b * L_ + (size_t)(q0 + m)) * H_ + h) * E_;
        Og[o + 0  + lm] = acc[0][v] * rinv;
        Og[o + 16 + lm] = acc[1][v] * rinv;
        Og[o + 32 + lm] = acc[2][v] * rinv;
        Og[o + 48 + lm] = acc[3][v] * rinv;
    }
}

extern "C" void kernel_launch(void* const* d_in, const int* in_sizes, int n_in,
                              void* d_out, int out_size, void* d_ws, size_t ws_size,
                              hipStream_t stream) {
    const float* Q = (const float*)d_in[0];
    const float* K = (const float*)d_in[1];
    const float* V = (const float*)d_in[2];
    // d_in[3] (attn_mask) ignored: causal mask computed analytically.
    float* O = (float*)d_out;

    const size_t nelem = (size_t)B_ * H_ * L_ * E_;      // 8,388,608
    _Float16* Kh = (_Float16*)d_ws;                      // 16.78 MB
    _Float16* Vt = Kh + nelem;                           // 16.78 MB

    conv_k_kernel <<<(nelem / 4) / 256, 256, 0, stream>>>(K, Kh);
    trans_v_kernel<<<B_ * H_ * (L_ / 64), 256, 0, stream>>>(V, Vt);

    fa_causal_wmma_kernel<<<B_ * H_ * (L_ / 128), 256, 0, stream>>>(Q, Kh, Vt, O);
}
